// StateSpaceSimulator_67972152426899
// MI455X (gfx1250) — compile-verified, matrix operand order checked by reading, not run on hardware
//
#include <hip/hip_runtime.h>
#include <hip/hip_bf16.h>

// StateSpaceSimulator for MI455X (gfx1250):
//  - 32 blocks x 256 threads (8 waves); each block owns 16 batch rows for all
//    2048 time steps (rows independent -> state resident on-chip).
//  - W1/W2 converted to f16 once, held as WMMA B-fragments in VGPRs.
//  - x in LDS (f32 master + f16 A-operand copy); u double-buffered in LDS so
//    staging u(t+1) and NT-storing x(t) overlap GEMM1 -> only 2 barriers/step.
//  - GEMM1 (K=96, 3 wmma chains x 2 N-tiles) + v_tanh_f32 -> h (LDS),
//    GEMM2 (K=256, 8 wmma), x += dx. Bias rides in as the first WMMA's C.

typedef _Float16 f16;
typedef __attribute__((ext_vector_type(16))) _Float16 v16h;
typedef __attribute__((ext_vector_type(8)))  _Float16 v8h;
typedef __attribute__((ext_vector_type(8)))  float    v8f;
typedef __attribute__((ext_vector_type(4)))  float    v4f;

#define N_BATCH 512
#define L_STEPS 2048
#define NX 64
#define NU 32
#define NF 256
#define TM 16      // batch rows per block (WMMA M)

union FragU { v16h v16; v8h v8[2]; f16 h[16]; };

__device__ __forceinline__ unsigned lane_id() { return threadIdx.x & 31u; }

// Branch-free tanh: gfx1250 V_TANH_F32 transcendental (confirmed in codegen).
__device__ __forceinline__ float fast_tanh(float x) {
#if __has_builtin(__builtin_amdgcn_tanhf)
  return __builtin_amdgcn_tanhf(x);
#else
  float e = __builtin_amdgcn_exp2f(fminf(x, 40.0f) * 2.885390081777927f);
  return (e - 1.0f) * __builtin_amdgcn_rcpf(e + 1.0f);
#endif
}

// Load a 16x32 f16 A-tile fragment from a row-major LDS matrix with leading
// dim `ld`, K-chunk starting at k0. Per ISA 7.12.2 (16-bit A 16x32):
//   lanes 0-15 : row M=lane,    K = {k0+0..7,  k0+16..23}
//   lanes 16-31: row M=lane-16, K = {k0+8..15, k0+24..31}
__device__ __forceinline__ v16h load_A(const f16* base, int ld, int k0) {
  unsigned l   = lane_id();
  unsigned row = l & 15u;
  unsigned ko  = (l >> 4) ? 8u : 0u;
  const f16* p = base + row * ld + k0 + ko;
  FragU f;
  f.v8[0] = *(const v8h*)(p);        // K = k0+ko .. +7      (16B LDS load)
  f.v8[1] = *(const v8h*)(p + 16);   // K = k0+ko+16 .. +23  (16B LDS load)
  return f.v16;
}

// Build a 32x16 B-tile fragment (column-per-lane, K split across lane halves,
// mirroring the A layout) directly from a row-major f32 weight W[K][N].
// Done ONCE before the time loop, so strided scalar loads are fine.
__device__ __forceinline__ v16h load_B_f32(const float* w, int ldn, int k0, int n0) {
  unsigned l   = lane_id();
  unsigned col = l & 15u;
  unsigned ko  = (l >> 4) ? 8u : 0u;
  FragU f;
#pragma unroll
  for (int i = 0; i < 8; ++i) {
    f.h[i]     = (f16)w[(k0 + ko + i)      * ldn + n0 + col];
    f.h[i + 8] = (f16)w[(k0 + ko + 16 + i) * ldn + n0 + col];
  }
  return f.v16;
}

__global__ __launch_bounds__(256)
void ssm_step_kernel(const float* __restrict__ x0, const float* __restrict__ u,
                     const float* __restrict__ W1, const float* __restrict__ b1,
                     const float* __restrict__ W2, const float* __restrict__ b2,
                     float* __restrict__ out) {
  __shared__ alignas(16) f16   s_xf[TM * NX];     // x as f16 (GEMM1 K=0..63)
  __shared__ alignas(16) f16   s_u [2][TM * NU];  // double-buffered u_t (K=64..95)
  __shared__ alignas(16) f16   s_h [TM * NF];     // hidden activations f16
  __shared__ alignas(16) float s_x [TM * NX];     // f32 master state

  const int tid  = threadIdx.x;
  const int wave = tid >> 5;
  const unsigned l     = tid & 31u;
  const unsigned col   = l & 15u;            // C/D column within 16-wide N tile
  const unsigned rhalf = (l >> 4) ? 8u : 0u; // C/D row offset (M = r + rhalf)
  const int row0 = blockIdx.x * TM;          // first batch row of this block

  // ---- Preload time-invariant weight B-fragments into VGPRs ----
  // GEMM1: wave w owns hidden cols [32w, 32w+32) -> 2 N-tiles x 3 K-chunks.
  const int g1n0 = wave * 32;
  v16h w1f[2][3];
#pragma unroll
  for (int nt = 0; nt < 2; ++nt)
#pragma unroll
    for (int kc = 0; kc < 3; ++kc)
      w1f[nt][kc] = load_B_f32(W1, NF, kc * 32, g1n0 + nt * 16);
  v8f bias1v[2];
#pragma unroll
  for (int nt = 0; nt < 2; ++nt) {
    const float bv = b1[g1n0 + nt * 16 + col];
#pragma unroll
    for (int i = 0; i < 8; ++i) bias1v[nt][i] = bv;
  }

  // GEMM2: waves 0-3 own state cols [16w, 16w+16) -> 8 K-chunks.
  const int g2n0 = (wave & 3) * 16;
  v16h w2f[8];
#pragma unroll
  for (int kc = 0; kc < 8; ++kc)
    w2f[kc] = load_B_f32(W2, NX, kc * 32, g2n0);
  v8f bias2v;
  {
    const float bv = b2[g2n0 + col];
#pragma unroll
    for (int i = 0; i < 8; ++i) bias2v[i] = bv;
  }

  // ---- Initialize state tile + stage u(0); start u(1) in flight ----
  const int tr = tid >> 4;             // 0..15: row handled by this thread
  const int tc = tid & 15;             // 0..15: column group
  {
    const v4f v = *(const v4f*)(x0 + (size_t)(row0 + tr) * NX + tc * 4);
    *(v4f*)(s_x + tr * NX + tc * 4) = v;
    s_xf[tr * NX + tc * 4 + 0] = (f16)v.x;
    s_xf[tr * NX + tc * 4 + 1] = (f16)v.y;
    s_xf[tr * NX + tc * 4 + 2] = (f16)v.z;
    s_xf[tr * NX + tc * 4 + 3] = (f16)v.w;
  }
  const float* ubase = u + (size_t)(row0 + tr) * L_STEPS * NU + tc * 2;
  {
    float2 u0 = *(const float2*)(ubase);
    union { f16 h[2]; unsigned b; } pk;
    pk.h[0] = (f16)u0.x;
    pk.h[1] = (f16)u0.y;
    *(unsigned*)(s_u[0] + tr * NU + tc * 2) = pk.b;
  }
  float2 uv = *(const float2*)(ubase + NU);   // u(1), consumed next iteration
  __syncthreads();

  for (int t = 0; t < L_STEPS; ++t) {
    // ---- Phase B: out-store x(t), stage u(t+1), GEMM1 + tanh -> h ----
    // (all reads of s_x/s_xf/s_u[t&1] here; writes go to out / s_u[(t+1)&1]/s_h)
    {
      v4f xv = *(const v4f*)(s_x + tr * NX + tc * 4);
      __builtin_nontemporal_store(
          xv, (v4f*)(out + ((size_t)(row0 + tr) * L_STEPS + t) * NX + tc * 4));
    }
    if (t + 1 < L_STEPS) {
      union { f16 h[2]; unsigned b; } pk;
      pk.h[0] = (f16)uv.x;
      pk.h[1] = (f16)uv.y;
      *(unsigned*)(s_u[(t + 1) & 1] + tr * NU + tc * 2) = pk.b;
      if (t + 2 < L_STEPS)
        uv = *(const float2*)(ubase + (size_t)(t + 2) * NU);  // hidden by step
      if (t + 16 < L_STEPS)
        __builtin_prefetch(ubase + (size_t)(t + 16) * NU, 0, 1);
    }

    {
      v16h a0 = load_A(s_xf, NX, 0);            // x cols  0..31
      v16h a1 = load_A(s_xf, NX, 32);           // x cols 32..63
      v16h a2 = load_A(s_u[t & 1], NU, 0);      // u cols  0..31
#pragma unroll
      for (int nt = 0; nt < 2; ++nt) {
        v8f acc = __builtin_amdgcn_wmma_f32_16x16x32_f16(
            false, a0, false, w1f[nt][0], (short)0, bias1v[nt], false, false);
        acc = __builtin_amdgcn_wmma_f32_16x16x32_f16(
            false, a1, false, w1f[nt][1], (short)0, acc, false, false);
        acc = __builtin_amdgcn_wmma_f32_16x16x32_f16(
            false, a2, false, w1f[nt][2], (short)0, acc, false, false);
        const int hc = g1n0 + nt * 16 + col;
#pragma unroll
        for (int i = 0; i < 8; ++i)
          s_h[(i + rhalf) * NF + hc] = (f16)fast_tanh(acc[i]);
      }
    }
    __syncthreads();   // h ready; all s_x/s_xf readers done

    // ---- Phase C: GEMM2 (waves 0-3), x += dx ----
    if (wave < 4) {                       // wave-uniform: EXEC stays all-1s
      v8f acc = __builtin_amdgcn_wmma_f32_16x16x32_f16(
          false, load_A(s_h, NF, 0), false, w2f[0], (short)0, bias2v, false, false);
#pragma unroll
      for (int kc = 1; kc < 8; ++kc) {
        v16h a = load_A(s_h, NF, kc * 32);
        acc = __builtin_amdgcn_wmma_f32_16x16x32_f16(
            false, a, false, w2f[kc], (short)0, acc, false, false);
      }
      const int xc = g2n0 + col;
#pragma unroll
      for (int i = 0; i < 8; ++i) {
        const int r = i + rhalf;
        float xn = s_x[r * NX + xc] + acc[i];
        s_x [r * NX + xc] = xn;           // f32 master state
        s_xf[r * NX + xc] = (f16)xn;      // f16 copy for next step's A-frag
      }
    }
    __syncthreads();   // x(t+1) and u(t+1) staged for next iteration
  }
}

extern "C" void kernel_launch(void* const* d_in, const int* in_sizes, int n_in,
                              void* d_out, int out_size, void* d_ws, size_t ws_size,
                              hipStream_t stream) {
  (void)in_sizes; (void)n_in; (void)out_size; (void)d_ws; (void)ws_size;
  const float* x0 = (const float*)d_in[0];
  const float* u  = (const float*)d_in[1];
  const float* W1 = (const float*)d_in[2];
  const float* b1 = (const float*)d_in[3];
  const float* W2 = (const float*)d_in[4];
  const float* b2 = (const float*)d_in[5];
  float* out = (float*)d_out;

  dim3 grid(N_BATCH / TM);   // 32 blocks
  dim3 block(256);           // 8 waves (wave32)
  ssm_step_kernel<<<grid, block, 0, stream>>>(x0, u, W1, b1, W2, b2, out);
}
